// Mixer_9251359555652
// MI455X (gfx1250) — matrix operand order repaired and checked by reading, not compiled, not run on hardware
//
#include <hip/hip_runtime.h>
#include <hip/hip_bf16.h>

// ---------------------------------------------------------------------------
// Mamba-2 / SSD mixer for MI455X (gfx1250, wave32, WMMA).
// All matmuls use v_wmma_f32_16x16x32_f16 (f32 accumulate).
// GEMMs are LDS double-buffered, branch-free staging, global_prefetch_b8.
// ---------------------------------------------------------------------------

typedef __attribute__((ext_vector_type(16))) _Float16 v16h;
typedef __attribute__((ext_vector_type(8)))  _Float16 half8;
typedef __attribute__((ext_vector_type(8)))  float    v8f;

#define BB      2
#define LL      4096
#define DMODEL  1024
#define DSTATE  64
#define NHEAD   32          // NV == NQK == 32 (rep == 1)
#define CHUNK   128
#define DINNER  1024
#define HDIM    32
#define CONVDIM 5120        // DINNER + 2*NHEAD*DSTATE
#define PROJ    6176        // 2*DINNER + 2*NHEAD*DSTATE + NHEAD
#define NC      (LL / CHUNK)   // 32 chunks
#define MTOK    (BB * LL)      // 8192 tokens

// ---------------- WMMA helpers --------------------------------------------

__device__ __forceinline__ v8f wmma_f16(v16h a, v16h b, v8f c) {
  return __builtin_amdgcn_wmma_f32_16x16x32_f16(
      /*neg_a=*/false, a, /*neg_b=*/false, b,
      /*c_mod=*/(short)0, c, /*reuse_a=*/false, /*reuse_b=*/false);
}

// A fragment (16xK tile, 16-bit): lanes 0-15 hold row M=lane with K in
// {kb..kb+15}; lanes 16-31 hold K in {kb+16..kb+31}.
__device__ __forceinline__ v16h ldsA(const _Float16* base, int ld, int row0, int kb) {
  int lane = threadIdx.x & 31;
  int row  = row0 + (lane & 15);
  int klo  = kb + ((lane >> 4) << 4);
  const _Float16* p = base + row * ld + klo;
  half8 lo = *(const half8*)(p);
  half8 hi = *(const half8*)(p + 8);
  v16h a;
#pragma unroll
  for (int i = 0; i < 8; ++i) { a[i] = lo[i]; a[i + 8] = hi[i]; }
  return a;
}

// B fragment (Kx16 tile, 16-bit), LDS stored column-major [col][k]:
// lane holds column N=lane&15, 16 contiguous K values starting at
// kb (lanes 0-15) / kb+16 (lanes 16-31).
__device__ __forceinline__ v16h ldsB(const _Float16* base, int ld, int col0, int kb) {
  int lane = threadIdx.x & 31;
  int col  = col0 + (lane & 15);
  int k0   = kb + ((lane >> 4) << 4);
  const _Float16* p = base + col * ld + k0;
  half8 lo = *(const half8*)(p);
  half8 hi = *(const half8*)(p + 8);
  v16h b;
#pragma unroll
  for (int i = 0; i < 8; ++i) { b[i] = lo[i]; b[i + 8] = hi[i]; }
  return b;
}

// Same fragments but sourced from row-major f32 global memory (convert inline).
__device__ __forceinline__ v16h gldA(const float* base, int ld, int row0, int kb) {
  int lane = threadIdx.x & 31;
  int row  = row0 + (lane & 15);
  int klo  = kb + ((lane >> 4) << 4);
  const float* p = base + (long)row * ld + klo;
  v16h a;
#pragma unroll
  for (int i = 0; i < 8; ++i) {
    a[i]     = (_Float16)p[i];
    a[i + 8] = (_Float16)p[i + 8];
  }
  return a;
}

__device__ __forceinline__ v16h gldB(const float* base, int ld, int col0, int kb) {
  int lane = threadIdx.x & 31;
  int col  = col0 + (lane & 15);
  int k0   = kb + ((lane >> 4) << 4);
  const float* p = base + (long)col * ld + k0;
  v16h b;
#pragma unroll
  for (int i = 0; i < 16; ++i) b[i] = (_Float16)p[i];
  return b;
}

// ---------------- Kernel 1: proj = u @ W_in  (8192x1024 @ 1024x6176) -------
// 256 threads = 8 waves, block tile 128x64, K step 32, f16 LDS staging,
// ping-pong double buffer (1 barrier / K-step, loads overlap WMMA).

__global__ __launch_bounds__(256) void k_gemm_proj(
    const float* __restrict__ A, const float* __restrict__ Bw,
    float* __restrict__ C, int M, int K, int N) {
  __shared__ __align__(16) _Float16 As[2][128 * 48];
  __shared__ __align__(16) _Float16 Bs[2][64 * 48];
  int m0 = blockIdx.x * 128;
  int n0 = blockIdx.y * 64;
  int tid = threadIdx.x;
  int wv = tid >> 5, wm = wv >> 1, wn = wv & 1;
  v8f acc00 = {}, acc01 = {}, acc10 = {}, acc11 = {};

  // per-thread staging coordinates (fixed across K loop)
  int r  = tid >> 1;
  int kc = (tid & 1) << 4;
  int kk = tid >> 3;
  int nq = (tid & 7) << 3;
  // N is a multiple of 8, n0 a multiple of 64 -> each 8-wide chunk is fully
  // in range or fully out.  Out-of-range chunks only feed output columns the
  // epilogue never stores, so clamp the chunk base instead of predicating.
  int nqc = (n0 + nq + 8 <= N) ? nq : 0;
  const float* ap = A + (long)(m0 + r) * K + kc;
  const float* bp = Bw + (long)kk * N + n0 + nqc;

  auto stage = [&](int buf, int kb) {
    const float* pa = ap + kb;
    _Float16* qa = As[buf] + r * 48 + kc;
#pragma unroll
    for (int i = 0; i < 16; ++i) qa[i] = (_Float16)pa[i];
    const float* pb = bp + (long)kb * N;
#pragma unroll
    for (int i = 0; i < 8; ++i) Bs[buf][(nq + i) * 48 + kk] = (_Float16)pb[i];
  };

  stage(0, 0);
  int buf = 0;
  for (int kb = 0; kb < K; kb += 32) {
    __syncthreads();
    if (kb + 32 < K) {
      __builtin_prefetch(ap + kb + 64, 0, 1);
      stage(buf ^ 1, kb + 32);
    }
    const _Float16* as = As[buf];
    const _Float16* bs = Bs[buf];
    v16h a0 = ldsA(as, 48, wm * 32,      0);
    v16h a1 = ldsA(as, 48, wm * 32 + 16, 0);
    v16h b0 = ldsB(bs, 48, wn * 32,      0);
    v16h b1 = ldsB(bs, 48, wn * 32 + 16, 0);
    acc00 = wmma_f16(a0, b0, acc00);
    acc01 = wmma_f16(a0, b1, acc01);
    acc10 = wmma_f16(a1, b0, acc10);
    acc11 = wmma_f16(a1, b1, acc11);
    buf ^= 1;
  }
  int lane = tid & 31;
  int p0 = lane & 15, hi = (lane >> 4) << 3;
#pragma unroll
  for (int rr = 0; rr < 8; ++rr) {
    int row0 = m0 + wm * 32 + hi + rr;
    int c0 = n0 + wn * 32 + p0;
    if (c0 < N)      C[(long)row0 * N + c0]             = acc00[rr];
    if (c0 + 16 < N) C[(long)row0 * N + c0 + 16]        = acc01[rr];
    if (c0 < N)      C[(long)(row0 + 16) * N + c0]      = acc10[rr];
    if (c0 + 16 < N) C[(long)(row0 + 16) * N + c0 + 16] = acc11[rr];
  }
}

// ---------------- Kernel 2: causal depthwise conv (k=4) + bias -------------

__global__ __launch_bounds__(256) void k_conv(
    const float* __restrict__ proj, const float* __restrict__ cw,
    const float* __restrict__ cb, float* __restrict__ xconv) {
  long idx = (long)blockIdx.x * blockDim.x + threadIdx.x;
  long total = (long)BB * LL * CONVDIM;
  if (idx >= total) return;
  int c = (int)(idx % CONVDIM);
  long bl = idx / CONVDIM;
  int l = (int)(bl % LL);
  long b = bl / LL;
  float acc = cb[c];
#pragma unroll
  for (int j = 0; j < 4; ++j) {
    int t = l - 3 + j;
    if (t >= 0) acc += proj[((long)b * LL + t) * PROJ + c] * cw[c * 4 + j];
  }
  xconv[idx] = acc;
}

// ---------------- Kernel 3: per-chunk cumsum of softplus(A_log) ------------

__global__ __launch_bounds__(128) void k_cum(
    const float* __restrict__ proj, float* __restrict__ cum) {
  __shared__ float s[128];
  int h  = blockIdx.x % NHEAD;
  int ch = (blockIdx.x / NHEAD) % NC;
  int b  = blockIdx.x / (NHEAD * NC);
  int t  = threadIdx.x;
  long l = (long)ch * CHUNK + t;
  float a = proj[((long)b * LL + l) * PROJ + (CONVDIM + DINNER) + h];
  float dt = (a > 20.f) ? a : log1pf(__expf(a));   // softplus
  s[t] = dt;
  __syncthreads();
  for (int off = 1; off < 128; off <<= 1) {
    float v = (t >= off) ? s[t - off] : 0.f;
    __syncthreads();
    s[t] += v;
    __syncthreads();
  }
  cum[((long)b * LL + l) * NHEAD + h] = s[t];
}

// ---------------- Kernel 4: S_local = B_cT @ (x * decay_last)  (64x32) -----

__global__ __launch_bounds__(128) void k_chunk_state(
    const float* __restrict__ xconv, const float* __restrict__ cum,
    float* __restrict__ Sloc) {
  __shared__ __align__(16) _Float16 Bt[64 * 136];  // [n][s]
  __shared__ __align__(16) _Float16 Xt[32 * 136];  // [p][s]
  __shared__ float cums[128];
  int h  = blockIdx.x % NHEAD;
  int ch = (blockIdx.x / NHEAD) % NC;
  int b  = blockIdx.x / (NHEAD * NC);
  long l0 = (long)ch * CHUNK;
  const float* base = xconv + ((long)b * LL + l0) * CONVDIM;
  const float* Bc = base + DINNER + h * DSTATE;
  const float* Xc = base + h * HDIM;
  int tid = threadIdx.x;
  cums[tid] = cum[((long)b * LL + l0 + tid) * NHEAD + h];
  __syncthreads();
  float clast = cums[127];
  for (int idx = tid; idx < 64 * 128; idx += 128) {   // Bt[n][s] = B_c[s][n]
    int s = idx >> 6, n = idx & 63;
    Bt[n * 136 + s] = (_Float16)Bc[(long)s * CONVDIM + n];
  }
  for (int idx = tid; idx < 32 * 128; idx += 128) {   // Xt[p][s] = x*decay
    int s = idx >> 5, p = idx & 31;
    float d = __expf(cums[s] - clast);
    Xt[p * 136 + s] = (_Float16)(Xc[(long)s * CONVDIM + p] * d);
  }
  __syncthreads();
  int wv = tid >> 5;
  v8f acc0 = {}, acc1 = {};
#pragma unroll
  for (int ks = 0; ks < 4; ++ks) {                    // K = 128 (time)
    v16h a = ldsA(Bt, 136, wv * 16, ks * 32);
    acc0 = wmma_f16(a, ldsB(Xt, 136, 0,  ks * 32), acc0);
    acc1 = wmma_f16(a, ldsB(Xt, 136, 16, ks * 32), acc1);
  }
  float* out = Sloc + (((long)(b * NC + ch)) * NHEAD + h) * (DSTATE * HDIM);
  int lane = tid & 31;
  int nb = wv * 16 + ((lane >> 4) << 3);
  int p0 = lane & 15;
#pragma unroll
  for (int rr = 0; rr < 8; ++rr) {
    out[(nb + rr) * HDIM + p0]      = acc0[rr];
    out[(nb + rr) * HDIM + 16 + p0] = acc1[rr];
  }
}

// ---------------- Kernel 5: sequential chunk scan (state recurrence) -------

__global__ __launch_bounds__(256) void k_scan(
    const float* __restrict__ Sloc, const float* __restrict__ cum,
    float* __restrict__ Sprev) {
  int h = blockIdx.x % NHEAD;
  int b = blockIdx.x / NHEAD;
  int tid = threadIdx.x;
  float S[8];
#pragma unroll
  for (int i = 0; i < 8; ++i) S[i] = 0.f;
  for (int c = 0; c < NC; ++c) {
    long off = (((long)(b * NC + c)) * NHEAD + h) * (DSTATE * HDIM);
    float tot = __expf(-cum[((long)b * LL + c * CHUNK + 127) * NHEAD + h]);
#pragma unroll
    for (int i = 0; i < 8; ++i) {
      int e = tid + i * 256;
      Sprev[off + e] = S[i];                    // state BEFORE this chunk
      S[i] = tot * S[i] + Sloc[off + e];
    }
  }
}

// ---------------- Kernel 6: y = exp(-cum)*(C@Sprev) + (G∘L)@x + D*x --------

__global__ __launch_bounds__(128) void k_chunk_out(
    const float* __restrict__ xconv, const float* __restrict__ cum,
    const float* __restrict__ Sprev, const float* __restrict__ Dp,
    float* __restrict__ Y) {
  __shared__ __align__(16) _Float16 Gs[128 * 136];  // masked G, f16
  __shared__ __align__(16) _Float16 Xsh[32 * 136];  // [p][s]
  __shared__ __align__(16) _Float16 Ssh[32 * 80];   // [p][n]
  __shared__ float cums[128];
  __shared__ float esc[128];
  int h  = blockIdx.x % NHEAD;
  int ch = (blockIdx.x / NHEAD) % NC;
  int b  = blockIdx.x / (NHEAD * NC);
  long l0 = (long)ch * CHUNK;
  const float* base = xconv + ((long)b * LL + l0) * CONVDIM;
  const float* Bc = base + DINNER + h * DSTATE;
  const float* Cc = base + DINNER + NHEAD * DSTATE + h * DSTATE;
  const float* Xc = base + h * HDIM;
  int tid = threadIdx.x, lane = tid & 31, wv = tid >> 5;

  cums[tid] = cum[((long)b * LL + l0 + tid) * NHEAD + h];
  __syncthreads();
  esc[tid] = __expf(-cums[tid]);

  // --- G = C @ B^T (128x128, K=64), masked + decay-scaled into LDS f16 ----
#pragma unroll
  for (int mt = 0; mt < 2; ++mt) {
    int row0 = wv * 32 + mt * 16;
    v16h a0 = gldA(Cc, CONVDIM, row0, 0);
    v16h a1 = gldA(Cc, CONVDIM, row0, 32);
    int tb = row0 + ((lane >> 4) << 3);
    for (int nt = 0; nt < 8; ++nt) {
      int s = nt * 16 + (lane & 15);
      if (nt * 16 > row0 + 15) {           // tile fully above the diagonal
#pragma unroll
        for (int rr = 0; rr < 8; ++rr) Gs[(tb + rr) * 136 + s] = (_Float16)0.f;
        continue;
      }
      v8f g = {};
      g = wmma_f16(a0, gldB(Bc, CONVDIM, nt * 16, 0),  g);
      g = wmma_f16(a1, gldB(Bc, CONVDIM, nt * 16, 32), g);
#pragma unroll
      for (int rr = 0; rr < 8; ++rr) {
        int t = tb + rr;
        float v = (s <= t) ? g[rr] * __expf(cums[s] - cums[t]) : 0.f;
        Gs[t * 136 + s] = (_Float16)v;
      }
    }
  }
  // stage x^T and S_prev^T
  for (int idx = tid; idx < 32 * 128; idx += 128) {
    int s = idx >> 5, p = idx & 31;
    Xsh[p * 136 + s] = (_Float16)Xc[(long)s * CONVDIM + p];
  }
  const float* Sp = Sprev + (((long)(b * NC + ch)) * NHEAD + h) * (DSTATE * HDIM);
  for (int idx = tid; idx < 64 * 32; idx += 128) {
    int n = idx >> 5, p = idx & 31;
    Ssh[p * 80 + n] = (_Float16)Sp[idx];
  }
  __syncthreads();

  // --- y tile (32 rows per wave, 32 cols) ---------------------------------
  float dph = Dp[h];
#pragma unroll
  for (int mt = 0; mt < 2; ++mt) {
    int row0 = wv * 32 + mt * 16;
    v8f acc0 = {}, acc1 = {};
#pragma unroll
    for (int ks = 0; ks < 2; ++ks) {       // y_inter: C @ Sprev (K=64)
      v16h a = gldA(Cc, CONVDIM, row0, ks * 32);
      acc0 = wmma_f16(a, ldsB(Ssh, 80, 0,  ks * 32), acc0);
      acc1 = wmma_f16(a, ldsB(Ssh, 80, 16, ks * 32), acc1);
    }
    int tb = row0 + ((lane >> 4) << 3);
#pragma unroll
    for (int rr = 0; rr < 8; ++rr) {       // row scale exp(-cum[t])
      float e = esc[tb + rr];
      acc0[rr] *= e; acc1[rr] *= e;
    }
#pragma unroll
    for (int ks = 0; ks < 4; ++ks) {       // y_intra: (G∘L) @ x (K=128)
      v16h a = ldsA(Gs, 136, row0, ks * 32);
      acc0 = wmma_f16(a, ldsB(Xsh, 136, 0,  ks * 32), acc0);
      acc1 = wmma_f16(a, ldsB(Xsh, 136, 16, ks * 32), acc1);
    }
    int p0 = lane & 15;
#pragma unroll
    for (int rr = 0; rr < 8; ++rr) {       // + Dp * x, store
      int t = tb + rr;
      long gro = ((long)b * LL + l0 + t) * DINNER + h * HDIM;
      float x0 = Xc[(long)t * CONVDIM + p0];
      float x1 = Xc[(long)t * CONVDIM + 16 + p0];
      Y[gro + p0]      = acc0[rr] + dph * x0;
      Y[gro + 16 + p0] = acc1[rr] + dph * x1;
    }
  }
}

// ---------------- Kernel 7: out = (y * silu(z+zb)) @ W_out -----------------
// Same double-buffered template; gating fused into the A-tile staging.

__global__ __launch_bounds__(256) void k_gemm_out(
    const float* __restrict__ Yin, const float* __restrict__ proj,
    const float* __restrict__ zb, const float* __restrict__ W,
    float* __restrict__ Out) {
  __shared__ __align__(16) _Float16 As[2][128 * 48];
  __shared__ __align__(16) _Float16 Bs[2][64 * 48];
  int m0 = blockIdx.x * 128;
  int n0 = blockIdx.y * 64;
  int tid = threadIdx.x;
  int wv = tid >> 5, wm = wv >> 1, wn = wv & 1;
  v8f acc00 = {}, acc01 = {}, acc10 = {}, acc11 = {};

  int r  = tid >> 1;
  int kc = (tid & 1) << 4;
  int kk = tid >> 3;
  int nq = (tid & 7) << 3;
  long row = m0 + r;
  const float* py0 = Yin  + row * (long)DINNER + kc;
  const float* pz0 = proj + row * (long)PROJ + CONVDIM + kc;
  const float* pw0 = W + (long)kk * DMODEL + n0 + nq;

  auto stage = [&](int buf, int kb) {
    const float* py = py0 + kb;
    const float* pz = pz0 + kb;
    _Float16* qa = As[buf] + r * 48 + kc;
#pragma unroll
    for (int i = 0; i < 16; ++i) {          // a = y * silu(z + z_bias)
      float z = pz[i] + zb[kb + kc + i];
      float sig = 1.f / (1.f + __expf(-z));
      qa[i] = (_Float16)(py[i] * z * sig);
    }
    const float* pw = pw0 + (long)kb * DMODEL;
#pragma unroll
    for (int i = 0; i < 8; ++i) Bs[buf][(nq + i) * 48 + kk] = (_Float16)pw[i];
  };

  stage(0, 0);
  int buf = 0;
  for (int kb = 0; kb < DINNER; kb += 32) {
    __syncthreads();
    if (kb + 32 < DINNER) {
      __builtin_prefetch(py0 + kb + 64, 0, 1);
      stage(buf ^ 1, kb + 32);
    }
    const _Float16* as = As[buf];
    const _Float16* bs = Bs[buf];
    v16h a0 = ldsA(as, 48, wm * 32,      0);
    v16h a1 = ldsA(as, 48, wm * 32 + 16, 0);
    v16h b0 = ldsB(bs, 48, wn * 32,      0);
    v16h b1 = ldsB(bs, 48, wn * 32 + 16, 0);
    acc00 = wmma_f16(a0, b0, acc00);
    acc01 = wmma_f16(a0, b1, acc01);
    acc10 = wmma_f16(a1, b0, acc10);
    acc11 = wmma_f16(a1, b1, acc11);
    buf ^= 1;
  }
  int lane = tid & 31;
  int p0 = lane & 15, hi = (lane >> 4) << 3;
#pragma unroll
  for (int rr = 0; rr < 8; ++rr) {
    int row0 = m0 + wm * 32 + hi + rr;
    int c0 = n0 + wn * 32 + p0;
    Out[(long)row0 * DMODEL + c0]             = acc00[rr];
    Out[(long)row0 * DMODEL + c0 + 16]        = acc01[rr];
    Out[(long)(row0 + 16) * DMODEL + c0]      = acc10[rr];
    Out[(long)(row0 + 16) * DMODEL + c0 + 16] = acc11[rr];
  }
}

// ---------------- Launch ---------------------------------------------------

extern "C" void kernel_launch(void* const* d_in, const int* in_sizes, int n_in,
                              void* d_out, int out_size, void* d_ws, size_t ws_size,
                              hipStream_t stream) {
  (void)in_sizes; (void)n_in; (void)out_size; (void)ws_size;
  const float* u      = (const float*)d_in[0];
  const float* W_in   = (const float*)d_in[1];
  const float* conv_w = (const float*)d_in[2];
  const float* conv_b = (const float*)d_in[3];
  const float* z_bias = (const float*)d_in[4];
  const float* Dp     = (const float*)d_in[5];
  const float* W_out  = (const float*)d_in[6];

  float* ws    = (float*)d_ws;
  float* proj  = ws;                                          // B*L*6176
  float* xconv = proj  + (size_t)BB * LL * PROJ;              // B*L*5120
  float* cum   = xconv + (size_t)BB * LL * CONVDIM;           // B*L*32
  float* Sloc  = cum   + (size_t)BB * LL * NHEAD;             // B*NC*NH*64*32
  float* Sprev = Sloc  + (size_t)BB * NC * NHEAD * DSTATE * HDIM;
  float* Yws   = Sprev + (size_t)BB * NC * NHEAD * DSTATE * HDIM;
  float* outp  = (float*)d_out;

  dim3 g1(MTOK / 128, (PROJ + 63) / 64);
  k_gemm_proj<<<g1, 256, 0, stream>>>(u, W_in, proj, MTOK, DMODEL, PROJ);

  long convN = (long)BB * LL * CONVDIM;
  k_conv<<<(unsigned)((convN + 255) / 256), 256, 0, stream>>>(proj, conv_w, conv_b, xconv);

  k_cum<<<BB * NC * NHEAD, 128, 0, stream>>>(proj, cum);
  k_chunk_state<<<BB * NC * NHEAD, 128, 0, stream>>>(xconv, cum, Sloc);
  k_scan<<<BB * NHEAD, 256, 0, stream>>>(Sloc, cum, Sprev);
  k_chunk_out<<<BB * NC * NHEAD, 128, 0, stream>>>(xconv, cum, Sprev, Dp, Yws);

  dim3 g2(MTOK / 128, DMODEL / 64);
  k_gemm_out<<<g2, 256, 0, stream>>>(Yws, proj, z_bias, W_out, outp);
}
